// SubjectSpecificProjection_49967649521824
// MI455X (gfx1250) — matrix operand order repaired and checked by reading, compiled, not run on hardware
//
#include <hip/hip_runtime.h>
#include <hip/hip_bf16.h>

#define NSUBJ    13
#define EEG_DIM  256
#define CLIP_DIM 512
#define BATCH    65536
#define TILE_M   128
#define TILE_N   128
#define TILE_K   32
#define MAX_TILES 526                 // ceil((BATCH + 13*127)/128) + slack
#define PERM_CAP  (MAX_TILES * 128)
#define LDS_STRIDE 40                 // padded row (ushorts) for bank-conflict-free frag reads

// ---- workspace layout (bytes) ----
#define WS_COUNTS   0
#define WS_CURSORS  64
#define WS_OFFSETS  128
#define WS_TILEDESC 256                      // int2 per tile: (subject, slotBase)
#define WS_PERM     4608                     // int[PERM_CAP]                  -> ends   273,920
#define WS_EEGB     274432                   // bf16[BATCH][EEG_DIM]           -> ends 33,828,864
#define WS_HBUF     33828864                 // bf16[PERM_CAP][CLIP_DIM]       -> ends 102,772,736
#define WS_W1P      102772736                // bf16 tiles [13][8][4][128][32] -> ends 106,180,608
#define WS_W2P      106180608                // bf16 tiles [13][16][4][128][32]-> ends 112,996,352

typedef __attribute__((ext_vector_type(16))) __bf16 v16bf;
typedef __attribute__((ext_vector_type(8)))  float  v8f;

union FragU { unsigned int u[8]; uint4 q[2]; v16bf v; };
union AccU  { v8f v; float f[8]; };

__device__ __forceinline__ unsigned int bf16pack2(float lo, float hi) {
  unsigned int a = __float_as_uint(lo), b = __float_as_uint(hi);
  a += 0x7FFFu + ((a >> 16) & 1u);           // RNE
  b += 0x7FFFu + ((b >> 16) & 1u);
  return (a >> 16) | (b & 0xFFFF0000u);
}
__device__ __forceinline__ unsigned short bf16s(float f) {
  unsigned int u = __float_as_uint(f);
  u += 0x7FFFu + ((u >> 16) & 1u);
  return (unsigned short)(u >> 16);
}

// ---------------- bucketing ----------------
__global__ void k_init(int* counts, int* cursors) {
  int t = threadIdx.x;
  if (t < 16) { counts[t] = 0; cursors[t] = 0; }
}

__global__ void k_count(const int* __restrict__ sub, int* counts, int* perm) {
  int i = blockIdx.x * blockDim.x + threadIdx.x;
  int stride = gridDim.x * blockDim.x;
  for (int t = i; t < BATCH; t += stride) atomicAdd(&counts[sub[t]], 1);
  for (int t = i; t < PERM_CAP; t += stride) perm[t] = -1;
}

__global__ void k_prefix(const int* counts, int* offsets, int* tileDesc) {
  if (threadIdx.x == 0 && blockIdx.x == 0) {
    int off = 0, t = 0;
    for (int s = 0; s < NSUBJ; s++) {
      offsets[s] = off;
      int padded = ((counts[s] + TILE_M - 1) / TILE_M) * TILE_M;
      for (int i = 0; i < padded / TILE_M; i++) {
        tileDesc[2 * t] = s; tileDesc[2 * t + 1] = off + i * TILE_M; t++;
      }
      off += padded;
    }
    for (; t < MAX_TILES; t++) { tileDesc[2 * t] = -1; tileDesc[2 * t + 1] = 0; }
  }
}

__global__ void k_scatter(const int* __restrict__ sub, const int* __restrict__ offsets,
                          int* cursors, int* perm) {
  int i = blockIdx.x * blockDim.x + threadIdx.x;
  int stride = gridDim.x * blockDim.x;
  for (int t = i; t < BATCH; t += stride) {
    int s = sub[t];
    int pos = atomicAdd(&cursors[s], 1);
    perm[offsets[s] + pos] = t;
  }
}

// ---------------- one-time precision/layout packing ----------------
// fp32 pairs -> packed bf16 (coalesced)
__global__ void k_cvt(const float* __restrict__ src, unsigned int* __restrict__ dst, int nPairs) {
  int i = blockIdx.x * blockDim.x + threadIdx.x;
  int stride = gridDim.x * blockDim.x;
  for (int t = i; t < nPairs; t += stride) {
    float2 f = ((const float2*)src)[t];
    dst[t] = bf16pack2(f.x, f.y);
  }
}

// W [13][K][512] fp32 -> bf16 tiles [s][k/32][nchunk(4)][n%128][k%32], tile = 4096 ushorts
__global__ void k_packW(const float* __restrict__ src, unsigned int* __restrict__ dst, int K) {
  int perS = (K / 32) * 4 * 128 * 16;       // uints per subject
  int total = NSUBJ * perS;
  int i = blockIdx.x * blockDim.x + threadIdx.x;
  int stride = gridDim.x * blockDim.x;
  for (int idx = i; idx < total; idx += stride) {
    int s  = idx / perS;   int r  = idx % perS;
    int kb = r / 8192;     int r2 = r % 8192;    // 4*128*16
    int nc = r2 / 2048;    int r3 = r2 % 2048;   // 128*16
    int nin = r3 / 16;     int kp = r3 % 16;
    int k = kb * 32 + kp * 2;
    int n = nc * 128 + nin;
    const float* p = src + (size_t)(s * K + k) * CLIP_DIM + n;
    dst[idx] = bf16pack2(p[0], p[CLIP_DIM]);
  }
}

// ---------------- grouped GEMM (bf16 WMMA, f32 acc) ----------------
// Tile: M=128 (8 waves x 16 rows), N=128 (8 sub-tiles of 16), K stepped by 32.
// A: bf16 rows of length KDIM (L1: eeg_bf16 gathered via perm; L2: hbuf by slot).
// B: pre-packed bf16 weight tiles, staged to LDS as a pure b128 copy.
template <int KDIM, bool L1>
__global__ __launch_bounds__(256) void gemm_kernel(
    const unsigned short* __restrict__ Abase,
    const unsigned short* __restrict__ Wp,
    const float* __restrict__ bias,
    const int* __restrict__ tileDesc, const int* __restrict__ perm,
    unsigned short* __restrict__ Hout, float* __restrict__ Yout) {
  __shared__ alignas(16) unsigned short Bt[TILE_N * LDS_STRIDE];

  int2 desc = ((const int2*)tileDesc)[blockIdx.x];
  int s = desc.x;
  if (s < 0) return;
  int slotBase = desc.y;
  int tid = threadIdx.x, lane = tid & 31, w = tid >> 5;

  // A-operand per-lane setup (16-bit A layout: lane m=l&15; k-half by l>>4)
  int kbaseA = (lane >> 4) * 8;
  int kbaseB = (lane >> 4) * 8;
  int ncol   = lane & 15;
  int slotA  = slotBase + w * 16 + (lane & 15);
  int rowIdx;
  if (L1) { int tok = perm[slotA]; rowIdx = tok < 0 ? 0 : tok; }
  else    { rowIdx = slotA; }
  const unsigned short* aRow = Abase + (size_t)rowIdx * KDIM;

  AccU acc[8];
#pragma unroll
  for (int i = 0; i < 8; i++)
#pragma unroll
    for (int j = 0; j < 8; j++) acc[i].f[j] = 0.0f;

  // staging indices: 256 threads copy a 4096-ushort tile (16 ushorts = 2x b128 each)
  int nin = tid >> 1, half = tid & 1;
  unsigned short* lrow = Bt + nin * LDS_STRIDE + half * 16;
  int g4idx = nin * 4 + half * 2;

  const size_t tileStride = 4096;           // ushorts per packed tile
  const unsigned short* gBase =
      Wp + ((size_t)(s * (KDIM / 32)) * 4 + blockIdx.y) * tileStride;

  for (int k0 = 0; k0 < KDIM; k0 += TILE_K) {
    __syncthreads();
    {
      const uint4* g4 = (const uint4*)(gBase + (size_t)(k0 >> 5) * 4 * tileStride);
      uint4 t0 = g4[g4idx], t1 = g4[g4idx + 1];
      ((uint4*)lrow)[0] = t0;
      *(uint4*)(lrow + 8) = t1;
      if (k0 + TILE_K < KDIM)
        __builtin_prefetch((const char*)(g4 + g4idx) + 4 * tileStride * 2, 0, 1);
    }
    __syncthreads();

    // A fragment: 16 bf16 per lane, two contiguous 16B halves
    FragU a;
    a.q[0] = *(const uint4*)(aRow + k0 + kbaseA);
    a.q[1] = *(const uint4*)(aRow + k0 + kbaseA + 16);

#pragma unroll
    for (int nS = 0; nS < 8; nS++) {
      FragU b;
      const unsigned short* bp = Bt + (nS * 16 + ncol) * LDS_STRIDE + kbaseB;
      b.q[0] = *(const uint4*)bp;
      b.q[1] = *(const uint4*)(bp + 16);
      acc[nS].v = __builtin_amdgcn_wmma_f32_16x16x32_bf16(
          false, a.v, false, b.v, (short)0, acc[nS].v, false, false);
    }
  }

  // epilogue (C layout: VGPR r -> M = r + 8*(lane>>4); N = lane&15)
  int nBase = blockIdx.y * TILE_N;
  int myM = (lane >> 4) * 8;
  if (L1) {
#pragma unroll
    for (int nS = 0; nS < 8; nS++) {
      int col = nBase + nS * 16 + ncol;
      float bv = bias[s * CLIP_DIM + col];
#pragma unroll
      for (int r = 0; r < 8; r++) {
        float v = acc[nS].f[r] + bv;
        v = v > 0.0f ? v : 0.0f;
        int slot = slotBase + w * 16 + myM + r;
        Hout[(size_t)slot * CLIP_DIM + col] = bf16s(v);
      }
    }
  } else {
    int toks[8];
#pragma unroll
    for (int r = 0; r < 8; r++) toks[r] = perm[slotBase + w * 16 + myM + r];
#pragma unroll
    for (int nS = 0; nS < 8; nS++) {
      int col = nBase + nS * 16 + ncol;
      float bv = bias[s * CLIP_DIM + col];
#pragma unroll
      for (int r = 0; r < 8; r++)
        if (toks[r] >= 0)
          Yout[(size_t)toks[r] * CLIP_DIM + col] = acc[nS].f[r] + bv;
    }
  }
}

// ---------------- row L2-normalize in place ----------------
__global__ __launch_bounds__(256) void k_norm(float* __restrict__ out) {
  int w = threadIdx.x >> 5, lane = threadIdx.x & 31;
  int rowsPerGrid = gridDim.x * 8;
  for (int row = blockIdx.x * 8 + w; row < BATCH; row += rowsPerGrid) {
    float* p = out + (size_t)row * CLIP_DIM + lane * 16;
    float4 a = ((float4*)p)[0], b = ((float4*)p)[1];
    float4 c = ((float4*)p)[2], d = ((float4*)p)[3];
    float ss = a.x*a.x + a.y*a.y + a.z*a.z + a.w*a.w
             + b.x*b.x + b.y*b.y + b.z*b.z + b.w*b.w
             + c.x*c.x + c.y*c.y + c.z*c.z + c.w*c.w
             + d.x*d.x + d.y*d.y + d.z*d.z + d.w*d.w;
    ss += __shfl_xor(ss, 16, 32);
    ss += __shfl_xor(ss, 8, 32);
    ss += __shfl_xor(ss, 4, 32);
    ss += __shfl_xor(ss, 2, 32);
    ss += __shfl_xor(ss, 1, 32);
    float nrm = sqrtf(ss);
    nrm = nrm > 1e-12f ? nrm : 1e-12f;
    float inv = 1.0f / nrm;
    a.x*=inv; a.y*=inv; a.z*=inv; a.w*=inv;
    b.x*=inv; b.y*=inv; b.z*=inv; b.w*=inv;
    c.x*=inv; c.y*=inv; c.z*=inv; c.w*=inv;
    d.x*=inv; d.y*=inv; d.z*=inv; d.w*=inv;
    ((float4*)p)[0] = a; ((float4*)p)[1] = b;
    ((float4*)p)[2] = c; ((float4*)p)[3] = d;
  }
}

extern "C" void kernel_launch(void* const* d_in, const int* in_sizes, int n_in,
                              void* d_out, int out_size, void* d_ws, size_t ws_size,
                              hipStream_t stream) {
  const float* eeg = (const float*)d_in[0];
  const int*   sub = (const int*)d_in[1];
  const float* W1  = (const float*)d_in[2];
  const float* b1  = (const float*)d_in[3];
  const float* W2  = (const float*)d_in[4];
  const float* b2  = (const float*)d_in[5];
  float* out = (float*)d_out;

  char* ws = (char*)d_ws;
  int* counts   = (int*)(ws + WS_COUNTS);
  int* cursors  = (int*)(ws + WS_CURSORS);
  int* offsets  = (int*)(ws + WS_OFFSETS);
  int* tileDesc = (int*)(ws + WS_TILEDESC);
  int* perm     = (int*)(ws + WS_PERM);
  unsigned short* eegB = (unsigned short*)(ws + WS_EEGB);
  unsigned short* hbuf = (unsigned short*)(ws + WS_HBUF);
  unsigned short* w1p  = (unsigned short*)(ws + WS_W1P);
  unsigned short* w2p  = (unsigned short*)(ws + WS_W2P);

  // routing tables
  k_init<<<1, 64, 0, stream>>>(counts, cursors);
  k_count<<<264, 256, 0, stream>>>(sub, counts, perm);
  k_prefix<<<1, 32, 0, stream>>>(counts, offsets, tileDesc);
  k_scatter<<<256, 256, 0, stream>>>(sub, offsets, cursors, perm);

  // one-time bf16 conversion / weight tiling
  k_cvt<<<4096, 256, 0, stream>>>(eeg, (unsigned int*)eegB, BATCH * EEG_DIM / 2);
  k_packW<<<2048, 256, 0, stream>>>(W1, (unsigned int*)w1p, EEG_DIM);
  k_packW<<<4096, 256, 0, stream>>>(W2, (unsigned int*)w2p, CLIP_DIM);

  dim3 grid(MAX_TILES, CLIP_DIM / TILE_N);
  gemm_kernel<EEG_DIM, true><<<grid, 256, 0, stream>>>(
      eegB, w1p, b1, tileDesc, perm, hbuf, nullptr);
  gemm_kernel<CLIP_DIM, false><<<grid, 256, 0, stream>>>(
      hbuf, w2p, b2, tileDesc, perm, nullptr, out);

  k_norm<<<2048, 256, 0, stream>>>(out);
}
